// DiGCN_62577673503439
// MI455X (gfx1250) — matrix operand order repaired and checked by reading, not compiled, not run on hardware
//
#include <hip/hip_runtime.h>
#include <hip/hip_bf16.h>

// ---------------------------------------------------------------------------
// GCN (DEPTH=2) on gfx1250: dual-GEMM via v_wmma_f32_16x16x32_f16,
// edge scatter via global_atomic_add_f32 (float4 per lane),
// BatchNorm via atomic column reduction.
// ---------------------------------------------------------------------------

typedef __attribute__((ext_vector_type(8)))  _Float16 v8h;
typedef __attribute__((ext_vector_type(16))) _Float16 v16h;
typedef __attribute__((ext_vector_type(8)))  float    v8f;

#define DD 128   // embed dim (fixed by reference)

// ---------------- generic helpers -----------------------------------------

__global__ __launch_bounds__(256)
void zero_f32(float* __restrict__ p, long long n) {
    long long i = (long long)blockIdx.x * blockDim.x + threadIdx.x;
    if (i < n) p[i] = 0.0f;
}

__global__ __launch_bounds__(256)
void cvt_f32_to_f16(const float* __restrict__ in, _Float16* __restrict__ out, long long n) {
    long long i = (long long)blockIdx.x * blockDim.x + threadIdx.x;
    if (i < n) out[i] = (_Float16)in[i];
}

// ---------------- degree / norm precompute ---------------------------------

__global__ __launch_bounds__(256)
void deg_kernel(const int* __restrict__ col, const float* __restrict__ w,
                float* __restrict__ deg, int E) {
    int e = blockIdx.x * blockDim.x + threadIdx.x;
    if (e < E) atomicAdd(&deg[col[e]], w[e]);
}

__global__ __launch_bounds__(256)
void dinv_kernel(float* __restrict__ deg, int n) {   // in-place deg -> deg^{-1/2}
    int i = blockIdx.x * blockDim.x + threadIdx.x;
    if (i < n) {
        float d = deg[i];
        deg[i] = (d > 0.0f) ? rsqrtf(d) : 0.0f;
    }
}

__global__ __launch_bounds__(256)
void norm_kernel(const int* __restrict__ row, const int* __restrict__ col,
                 const float* __restrict__ w, const float* __restrict__ dinv,
                 float* __restrict__ norm, int E) {
    int e = blockIdx.x * blockDim.x + threadIdx.x;
    if (e < E) norm[e] = dinv[row[e]] * w[e] * dinv[col[e]];
}

// ---------------- fused dual GEMM (WMMA f16 -> f32) ------------------------
// outg = h @ Wg^T,  outl = h @ Wl^T    (h:[N,128] f16 row-major, W:[128,128])
// Block: 256 threads = 8 waves; block owns 16 rows; wave w owns cols [16w,16w+16).
// A frag (16x32 f16, contiguous k order, same in B): lane l -> row l&15; half=l>>4;
//   v0..7 = K[kb+16*half .. +16)                               -> two 16B loads.
// B frag (32x16 f16): lane l -> col n=l&15 (= weight row n);
//   v0..7 = W[n][kb+16*half .. +16)                            -> one 32B load.
// C/D (16x16 f32): lane l holds col l&15; VGPR v holds row 8*(l>>4)+v.
__global__ __launch_bounds__(256)
void gemm_dual_f16(const _Float16* __restrict__ h16,
                   const _Float16* __restrict__ wg,
                   const _Float16* __restrict__ wl,
                   float* __restrict__ outg,
                   float* __restrict__ outl) {
    const int lane = threadIdx.x & 31;
    const int wave = threadIdx.x >> 5;
    const int l15  = lane & 15;
    const int hs   = lane >> 4;
    const int rowBase = blockIdx.x << 4;
    const int colBase = wave << 4;

    const _Float16* __restrict__ arow = h16 + (long long)(rowBase + l15) * DD;
    const _Float16* __restrict__ bgrow = wg + (colBase + l15) * DD;
    const _Float16* __restrict__ blrow = wl + (colBase + l15) * DD;

    v8f cg = {};
    v8f cl = {};

#pragma unroll
    for (int k = 0; k < DD; k += 32) {
        v8h a0 = *(const v8h*)(arow + k + 16 * hs);
        v8h a1 = *(const v8h*)(arow + k + 16 * hs + 8);
        v16h a;
#pragma unroll
        for (int i = 0; i < 8; ++i) { a[i] = a0[i]; a[i + 8] = a1[i]; }

        v16h Bg = *(const v16h*)(bgrow + k + 16 * hs);
        v16h Bl = *(const v16h*)(blrow + k + 16 * hs);

        cg = __builtin_amdgcn_wmma_f32_16x16x32_f16(false, a, false, Bg,
                                                    (short)0, cg, false, false);
        cl = __builtin_amdgcn_wmma_f32_16x16x32_f16(false, a, false, Bl,
                                                    (short)0, cl, false, false);
    }

    const int m0 = rowBase + 8 * hs;
    const int c  = colBase + l15;
#pragma unroll
    for (int v = 0; v < 8; ++v) {
        outg[(long long)(m0 + v) * DD + c] = cg[v];
        outl[(long long)(m0 + v) * DD + c] = cl[v];
    }
}

// ---------------- edge aggregation: hacc[col] += norm * hg[row] ------------
// 32 lanes per edge, float4 per lane (16 B vector load, 4 f32 atomics).
__global__ __launch_bounds__(256)
void agg_edges(const int* __restrict__ row, const int* __restrict__ col,
               const float* __restrict__ norm, const float* __restrict__ hg,
               float* __restrict__ hacc, int E) {
    long long gid = (long long)blockIdx.x * blockDim.x + threadIdx.x;
    int e = (int)(gid >> 5);
    if (e >= E) return;
    int lane = (int)(gid & 31);
    float nm = norm[e];
    const float4 v = *((const float4*)(hg + (long long)row[e] * DD) + lane);
    float* dst = hacc + (long long)col[e] * DD + lane * 4;
    atomicAdd(dst + 0, nm * v.x);
    atomicAdd(dst + 1, nm * v.y);
    atomicAdd(dst + 2, nm * v.z);
    atomicAdd(dst + 3, nm * v.w);
}

// ---------------- BatchNorm ------------------------------------------------

__global__ __launch_bounds__(256)
void bn_stats(const float* __restrict__ h, float* __restrict__ sum,
              float* __restrict__ sumsq, int nRows) {
    int gid    = blockIdx.x * blockDim.x + threadIdx.x;
    int c      = gid & (DD - 1);
    int r0     = gid >> 7;
    int stride = (gridDim.x * blockDim.x) >> 7;
    float s = 0.0f, ss = 0.0f;
    for (int r = r0; r < nRows; r += stride) {
        float v = h[(long long)r * DD + c];
        s += v;
        ss += v * v;
    }
    atomicAdd(&sum[c], s);
    atomicAdd(&sumsq[c], ss);
}

__global__ __launch_bounds__(128)
void bn_finalize(const float* __restrict__ sum, const float* __restrict__ sumsq,
                 float* __restrict__ mean, float* __restrict__ invstd,
                 float invN, float eps) {
    int c = threadIdx.x;
    float m = sum[c] * invN;
    float var = sumsq[c] * invN - m * m;
    mean[c]   = m;
    invstd[c] = rsqrtf(var + eps);
}

__global__ __launch_bounds__(256)
void bn_apply(const float* __restrict__ h,
              const float* __restrict__ mean, const float* __restrict__ invstd,
              const float* __restrict__ gamma, const float* __restrict__ beta,
              float* __restrict__ outF, _Float16* __restrict__ outH,
              int relu, long long total) {
    long long gid = (long long)blockIdx.x * blockDim.x + threadIdx.x;
    if (gid >= total) return;
    int c = (int)(gid & (DD - 1));
    float v = (h[gid] - mean[c]) * invstd[c] * gamma[c] + beta[c];
    if (relu) v = fmaxf(v, 0.0f);
    if (outF) outF[gid] = v;
    if (outH) outH[gid] = (_Float16)v;
}

// ---------------------------------------------------------------------------

extern "C" void kernel_launch(void* const* d_in, const int* in_sizes, int n_in,
                              void* d_out, int out_size, void* d_ws, size_t ws_size,
                              hipStream_t stream) {
    const float* x     = (const float*)d_in[0];
    const int*   eidx  = (const int*)d_in[1];
    const float* ew    = (const float*)d_in[2];
    const float* lin_w = (const float*)d_in[3];
    const float* gcn_w = (const float*)d_in[4];
    const float* gamma = (const float*)d_in[5];
    const float* beta  = (const float*)d_in[6];

    const int N = in_sizes[0] / DD;          // 100000
    const int E = in_sizes[1] / 2;           // 500000
    const int DEPTH = in_sizes[3] / (DD * DD); // 2
    const float EPS = 1e-5f;

    const int* row = eidx;
    const int* col = eidx + E;

    const long long Nd = (long long)N * DD;

    // ---- workspace carve-out (256B aligned slabs) ----
    char* ws = (char*)d_ws;
    size_t off = 0;
    auto carve = [&](size_t bytes) -> char* {
        char* p = ws + off;
        off += (bytes + 255) & ~(size_t)255;
        return p;
    };
    _Float16* h16   = (_Float16*)carve(Nd * sizeof(_Float16));
    float*    hg    = (float*)carve(Nd * sizeof(float));
    float*    hacc  = (float*)carve(Nd * sizeof(float));
    float*    nrm   = (float*)carve((size_t)E * sizeof(float));
    float*    dinv  = (float*)carve((size_t)N * sizeof(float));
    _Float16* wg16  = (_Float16*)carve((size_t)DD * DD * sizeof(_Float16));
    _Float16* wl16  = (_Float16*)carve((size_t)DD * DD * sizeof(_Float16));
    float*    stats = (float*)carve(4 * DD * sizeof(float)); // sum|sumsq|mean|invstd
    float* sum    = stats;
    float* sumsq  = stats + DD;
    float* meanv  = stats + 2 * DD;
    float* invstd = stats + 3 * DD;
    (void)ws_size; (void)n_in; (void)out_size;

    const int T = 256;
    auto nb = [](long long n, int t) { return (unsigned)((n + t - 1) / t); };

    // ---- gcn_norm: deg -> deg^{-1/2} -> per-edge norm ----
    zero_f32<<<nb(N, T), T, 0, stream>>>(dinv, N);
    deg_kernel<<<nb(E, T), T, 0, stream>>>(col, ew, dinv, E);
    dinv_kernel<<<nb(N, T), T, 0, stream>>>(dinv, N);
    norm_kernel<<<nb(E, T), T, 0, stream>>>(row, col, ew, dinv, nrm, E);

    // ---- initial f16 conversion of x ----
    cvt_f32_to_f16<<<nb(Nd, T), T, 0, stream>>>(x, h16, Nd);

    for (int l = 0; l < DEPTH; ++l) {
        // weights for this layer -> f16
        cvt_f32_to_f16<<<nb(DD * DD, T), T, 0, stream>>>(gcn_w + (size_t)l * DD * DD, wg16, DD * DD);
        cvt_f32_to_f16<<<nb(DD * DD, T), T, 0, stream>>>(lin_w + (size_t)l * DD * DD, wl16, DD * DD);

        // hg = h @ gcn_w^T ; hacc = h @ lin_w^T   (fused, one A-load)
        gemm_dual_f16<<<(unsigned)(N / 16), 256, 0, stream>>>(h16, wg16, wl16, hg, hacc);

        // hacc[col] += norm * hg[row]
        agg_edges<<<nb((long long)E * 32, T), T, 0, stream>>>(row, col, nrm, hg, hacc, E);

        // BatchNorm stats
        zero_f32<<<1, 256, 0, stream>>>(sum, 2 * DD);   // clears sum+sumsq (contiguous)
        bn_stats<<<256, 256, 0, stream>>>(hacc, sum, sumsq, N);
        bn_finalize<<<1, DD, 0, stream>>>(sum, sumsq, meanv, invstd, 1.0f / (float)N, EPS);

        const bool last = (l == DEPTH - 1);
        bn_apply<<<nb(Nd, T), T, 0, stream>>>(
            hacc, meanv, invstd, gamma + (size_t)l * DD, beta + (size_t)l * DD,
            last ? (float*)d_out : nullptr,
            last ? nullptr : h16,
            last ? 0 : 1, Nd);
    }
}
